// BasisVQ_19868518711964
// MI455X (gfx1250) — compile-verified, matrix operand order checked by reading, not run on hardware
//
#include <hip/hip_runtime.h>
#include <math.h>

typedef __attribute__((ext_vector_type(16))) __bf16 v16bf;
typedef __attribute__((ext_vector_type(8)))  __bf16 v8bf;
typedef __attribute__((ext_vector_type(8)))  float  v8f;
typedef __attribute__((ext_vector_type(4)))  float  v4f;

// Problem sizes (fixed by the reference)
constexpr int NROW = 16384;   // B*K = 256*64
constexpr int DIN  = 256;     // D
constexpr int DB   = 512;     // basis dim
constexpr int NC   = 1024;    // codebook size
constexpr int KTOP = 8;
constexpr float INV_TEMP = 10.0f;   // 1/0.1

// d_out layout (floats): q_st [NROW*DB] | indices [NROW] | vq_loss | entropy
constexpr int QSZ = NROW * DB;        // 8388608

__device__ __forceinline__ void top8_insert(float* tv, int* ti, float v, int id) {
    if (v >= tv[KTOP - 1]) return;
    int j = KTOP - 1;
    while (j > 0 && tv[j - 1] > v) { tv[j] = tv[j - 1]; ti[j] = ti[j - 1]; --j; }
    tv[j] = v; ti[j] = id;
}

__device__ __forceinline__ v16bf load_bf_frag(const __bf16* p) {
    // 8 bf16 (k..k+7) and 8 bf16 (k+16..k+23): two 16-byte loads
    v8bf lo = *(const v8bf*)(p);
    v8bf hi = *(const v8bf*)(p + 16);
    v16bf f;
#pragma unroll
    for (int i = 0; i < 8; ++i) { f[i] = lo[i]; f[8 + i] = hi[i]; }
    return f;
}

__device__ __forceinline__ v16bf load_f32_frag(const float* p) {
    v4f a0 = *(const v4f*)(p);
    v4f a1 = *(const v4f*)(p + 4);
    v4f b0 = *(const v4f*)(p + 16);
    v4f b1 = *(const v4f*)(p + 20);
    v16bf f;
#pragma unroll
    for (int i = 0; i < 4; ++i) {
        f[i]      = (__bf16)a0[i];
        f[4 + i]  = (__bf16)a1[i];
        f[8 + i]  = (__bf16)b0[i];
        f[12 + i] = (__bf16)b1[i];
    }
    return f;
}

// ---------------- Prep kernels ----------------

__global__ void k_zero(float* usage, float* lossacc) {
    int t = blockIdx.x * blockDim.x + threadIdx.x;
    if (t < NC) usage[t] = 0.0f;
    if (t == 0) lossacc[0] = 0.0f;
}

// wt[n][k] = (bf16) W[k][n]  (so WMMA B fragments are contiguous in k)
__global__ void k_wt_bf16(const float* __restrict__ W, __bf16* __restrict__ wt) {
    int i = blockIdx.x * 256 + threadIdx.x;       // over DB*DIN
    int n = i / DIN;
    int k = i - n * DIN;
    wt[i] = (__bf16)W[k * DB + n];
}

// One row (DB=512 cols) per block: convert fp32 -> bf16 and compute ||row||^2
__global__ void k_row_bf16_norm(const float* __restrict__ src,
                                __bf16* __restrict__ dst,
                                float* __restrict__ nrm) {
    __shared__ float red[256];
    int row = blockIdx.x;
    int t = threadIdx.x;
    const float* s = src + row * DB;
    float a = s[t], b = s[t + 256];
    dst[row * DB + t]       = (__bf16)a;
    dst[row * DB + t + 256] = (__bf16)b;
    red[t] = a * a + b * b;
    __syncthreads();
    for (int st = 128; st > 0; st >>= 1) {
        if (t < st) red[t] += red[t + st];
        __syncthreads();
    }
    if (t == 0) nrm[row] = red[0];
}

// ---------------- Projection GEMM: z = slots @ W + b ----------------
// grid (NROW/128, DB/64), block 256 = 8 waves; wave = 16 rows x 64 cols (4 tiles)
__global__ void __launch_bounds__(256) k_proj_wmma(const float* __restrict__ slots,
                                                   const __bf16* __restrict__ wt,
                                                   const float* __restrict__ bias,
                                                   float* __restrict__ z) {
    const int lane = threadIdx.x & 31;
    const int wv   = threadIdx.x >> 5;
    const int half = lane >> 4;
    const int l16  = lane & 15;
    const int koff = half ? 8 : 0;
    const int row0 = blockIdx.x * 128 + wv * 16;
    const int col0 = blockIdx.y * 64;

    v8f acc[4];
#pragma unroll
    for (int t = 0; t < 4; ++t)
#pragma unroll
        for (int i = 0; i < 8; ++i) acc[t][i] = 0.0f;

    const float* arow = slots + (row0 + l16) * DIN;
    for (int kb = 0; kb < DIN; kb += 32) {
        v16bf af = load_f32_frag(arow + kb + koff);
#pragma unroll
        for (int t = 0; t < 4; ++t) {
            const __bf16* bp = wt + (col0 + t * 16 + l16) * DIN + kb + koff;
            v16bf bf = load_bf_frag(bp);
            acc[t] = __builtin_amdgcn_wmma_f32_16x16x32_bf16(
                false, af, false, bf, (short)0, acc[t], false, false);
        }
    }
#pragma unroll
    for (int t = 0; t < 4; ++t) {
        int col = col0 + t * 16 + l16;
        float bv = bias[col];
#pragma unroll
        for (int r = 0; r < 8; ++r) {
            int row = row0 + r + half * 8;
            z[row * DB + col] = acc[t][r] + bv;
        }
    }
}

// ---------------- Fused distance + top-8 + softmax + gather ----------------
// grid NROW/16, block 256 = 8 waves; wave owns 128 of the 1024 codes.
__global__ void __launch_bounds__(256) k_dist_topk(const __bf16* __restrict__ zb,
                                                   const __bf16* __restrict__ eb,
                                                   const float* __restrict__ znorm,
                                                   const float* __restrict__ enorm,
                                                   const float* __restrict__ zf,
                                                   const float* __restrict__ embed,
                                                   float* __restrict__ usage,
                                                   float* __restrict__ lossacc,
                                                   float* __restrict__ out) {
    __shared__ float s_dist[16][NC];           // 64 KB (WGP has 320 KB LDS)
    __shared__ float s_cv[16][16][KTOP];       // 8 KB
    __shared__ int   s_ci[16][16][KTOP];       // 8 KB
    __shared__ float s_wv[16][KTOP];
    __shared__ int   s_wi[16][KTOP];
    __shared__ float s_red[256];

    const int tid  = threadIdx.x;
    const int lane = tid & 31;
    const int wv   = tid >> 5;
    const int half = lane >> 4;
    const int l16  = lane & 15;
    const int koff = half ? 8 : 0;
    const int row0 = blockIdx.x * 16;

    v8f acc[8];
#pragma unroll
    for (int t = 0; t < 8; ++t)
#pragma unroll
        for (int i = 0; i < 8; ++i) acc[t][i] = 0.0f;

    // dot = z @ embed^T  (embed rows are the B columns: already k-contiguous)
    const __bf16* arow = zb + (row0 + l16) * DB;
    for (int kb = 0; kb < DB; kb += 32) {
        v16bf af = load_bf_frag(arow + kb + koff);
#pragma unroll
        for (int t = 0; t < 8; ++t) {
            const __bf16* bp = eb + (wv * 128 + t * 16 + l16) * DB + kb + koff;
            v16bf bf = load_bf_frag(bp);
            acc[t] = __builtin_amdgcn_wmma_f32_16x16x32_bf16(
                false, af, false, bf, (short)0, acc[t], false, false);
        }
    }

    // dist = sqrt(max(|z|^2 + |e|^2 - 2 dot, 0)) -> LDS
#pragma unroll
    for (int t = 0; t < 8; ++t) {
        int col = wv * 128 + t * 16 + l16;
        float en = enorm[col];
#pragma unroll
        for (int r = 0; r < 8; ++r) {
            int m = r + half * 8;
            float d2 = znorm[row0 + m] + en - 2.0f * acc[t][r];
            s_dist[m][col] = sqrtf(fmaxf(d2, 0.0f));
        }
    }
    __syncthreads();

    // stage-1 top-8: 16 threads per row scan strided slices
    {
        int rr = tid >> 4;
        int sl = tid & 15;
        float tv[KTOP]; int ti[KTOP];
#pragma unroll
        for (int j = 0; j < KTOP; ++j) { tv[j] = 3.4e38f; ti[j] = 0; }
        for (int c = sl; c < NC; c += 16) top8_insert(tv, ti, s_dist[rr][c], c);
#pragma unroll
        for (int j = 0; j < KTOP; ++j) { s_cv[rr][sl][j] = tv[j]; s_ci[rr][sl][j] = ti[j]; }
    }
    __syncthreads();

    // stage-2 merge + softmax(-dist/T); write index output; accumulate usage
    if (tid < 16) {
        float tv[KTOP]; int ti[KTOP];
#pragma unroll
        for (int j = 0; j < KTOP; ++j) { tv[j] = 3.4e38f; ti[j] = 0; }
        for (int s = 0; s < 16; ++s)
#pragma unroll
            for (int j = 0; j < KTOP; ++j)
                top8_insert(tv, ti, s_cv[tid][s][j], s_ci[tid][s][j]);
        float mn = tv[0];                 // sorted ascending
        float w[KTOP]; float tot = 0.0f;
#pragma unroll
        for (int j = 0; j < KTOP; ++j) { w[j] = expf(-(tv[j] - mn) * INV_TEMP); tot += w[j]; }
        float inv = 1.0f / tot;
#pragma unroll
        for (int j = 0; j < KTOP; ++j) {
            w[j] *= inv;
            s_wv[tid][j] = w[j];
            s_wi[tid][j] = ti[j];
            atomicAdd(&usage[ti[j]], w[j]);
        }
        out[QSZ + row0 + tid] = (float)ti[0];
    }
    __syncthreads();

    // q = sum_j w_j * embed[idx_j]; loss partial = sum (z - q)^2
    float lsum = 0.0f;
    for (int e = tid; e < 16 * DB; e += 256) {
        int m = e >> 9;
        int c = e & (DB - 1);
        float q = 0.0f;
#pragma unroll
        for (int j = 0; j < KTOP; ++j) q += s_wv[m][j] * embed[s_wi[m][j] * DB + c];
        out[(row0 + m) * DB + c] = q;
        float diff = zf[(row0 + m) * DB + c] - q;
        lsum += diff * diff;
    }
    s_red[tid] = lsum;
    __syncthreads();
    for (int st = 128; st > 0; st >>= 1) {
        if (tid < st) s_red[tid] += s_red[tid + st];
        __syncthreads();
    }
    if (tid == 0) atomicAdd(lossacc, s_red[0]);
}

// ---------------- Finalize: entropy + scalar outputs ----------------
__global__ void k_finalize(const float* __restrict__ usage,
                           const float* __restrict__ lossacc,
                           float* __restrict__ out) {
    __shared__ float red[NC];
    int t = threadIdx.x;
    float u = usage[t] * (1.0f / (float)NROW);
    red[t] = u * logf(u + 1e-8f);
    __syncthreads();
    for (int st = 512; st > 0; st >>= 1) {
        if (t < st) red[t] += red[t + st];
        __syncthreads();
    }
    if (t == 0) {
        out[QSZ + NROW]     = lossacc[0] / ((float)NROW * (float)DB);
        out[QSZ + NROW + 1] = -red[0];
    }
}

// ---------------- Host launch ----------------
extern "C" void kernel_launch(void* const* d_in, const int* in_sizes, int n_in,
                              void* d_out, int out_size, void* d_ws, size_t ws_size,
                              hipStream_t stream) {
    const float* slots = (const float*)d_in[0];   // [16384, 256]
    const float* projW = (const float*)d_in[1];   // [256, 512]
    const float* projb = (const float*)d_in[2];   // [512]
    const float* embed = (const float*)d_in[3];   // [1024, 512]
    float* out = (float*)d_out;

    char* ws = (char*)d_ws;
    float*  z_f32   = (float*)(ws);                          // 32 MB
    __bf16* z_bf    = (__bf16*)(ws + 33554432);              // 16 MB
    __bf16* e_bf    = (__bf16*)(ws + 50331648);              // 1 MB
    __bf16* wt_bf   = (__bf16*)(ws + 51380224);              // 256 KB
    float*  enorm   = (float*)(ws + 51642368);               // 4 KB
    float*  znorm   = (float*)(ws + 51646464);               // 64 KB
    float*  usage   = (float*)(ws + 51712000);               // 4 KB
    float*  lossacc = (float*)(ws + 51716096);               // 4 B

    k_zero<<<4, 256, 0, stream>>>(usage, lossacc);
    k_wt_bf16<<<(DB * DIN) / 256, 256, 0, stream>>>(projW, wt_bf);
    k_row_bf16_norm<<<NC, 256, 0, stream>>>(embed, e_bf, enorm);

    dim3 pgrid(NROW / 128, DB / 64);
    k_proj_wmma<<<pgrid, 256, 0, stream>>>(slots, wt_bf, projb, z_f32);

    k_row_bf16_norm<<<NROW, 256, 0, stream>>>(z_f32, z_bf, znorm);

    k_dist_topk<<<NROW / 16, 256, 0, stream>>>(z_bf, e_bf, znorm, enorm,
                                               z_f32, embed, usage, lossacc, out);

    k_finalize<<<1, NC, 0, stream>>>(usage, lossacc, out);
}
